// SelfAttention_43533788512405
// MI455X (gfx1250) — matrix operand order repaired and checked by reading, not compiled, not run on hardware
//
#include <hip/hip_runtime.h>

// ---------------------------------------------------------------------------
// Self-attention forward for MI455X (gfx1250, wave32, WMMA f16 16x16x32).
// Pipeline: (1) QKV GEMM fp32->f16, writing Q,K row-major and V transposed
//           (2) fused flash attention per (b,h): K/V tiles staged in LDS via
//               async global->LDS copies, online softmax, P bounced via LDS
//           (3) projection GEMM + bias -> fp32 output
// Workspace layout (f16): Q[32][2048][64] | K[32][2048][64] | Vt[32][64][2048]
//                         | AO[4096][1024]   (4 * 8 MB = 32 MB)
// ---------------------------------------------------------------------------

typedef _Float16 h16;
typedef _Float16 v8h  __attribute__((ext_vector_type(8)));
typedef _Float16 v16h __attribute__((ext_vector_type(16)));
typedef float    v4f  __attribute__((ext_vector_type(4)));
typedef float    v8f  __attribute__((ext_vector_type(8)));
typedef int      v4i  __attribute__((ext_vector_type(4)));

#define DIM_C   1024
#define NHEADS  16
#define HD      64
#define BATCH   2
#define SEQ     2048
#define MTOT    (BATCH * SEQ)        // 4096
#define QK_SCALE 0.125f              // 64^-0.5 (exact power of two)

#define AS_GLOBAL __attribute__((address_space(1)))
#define AS_LDS    __attribute__((address_space(3)))

// --- async global->LDS staging (CDNA5 GLOBAL_LOAD_ASYNC_TO_LDS, ASYNCcnt) ---
#ifndef USE_ASYNC_LDS
#define USE_ASYNC_LDS 1
#endif

#if USE_ASYNC_LDS && defined(__has_builtin) && \
    __has_builtin(__builtin_amdgcn_global_load_async_to_lds_b128)
static __device__ __forceinline__ void async_cp_b128(const h16* g, h16* l) {
  __builtin_amdgcn_global_load_async_to_lds_b128(
      (AS_GLOBAL v4i*)g, (AS_LDS v4i*)l, /*offset=*/0, /*cpol=*/0);
}
static __device__ __forceinline__ void async_wait_all() {
#if __has_builtin(__builtin_amdgcn_s_wait_asynccnt)
  __builtin_amdgcn_s_wait_asynccnt(0);
#else
  asm volatile("s_wait_asynccnt 0" ::: "memory");
#endif
}
#else
static __device__ __forceinline__ void async_cp_b128(const h16* g, h16* l) {
  *(v8h*)l = *(const v8h*)g;
}
static __device__ __forceinline__ void async_wait_all() {}
#endif

static __device__ __forceinline__ v8f wmma_f16(v16h a, v16h b, v8f c) {
  // D = A(16x32 f16) * B(32x16 f16) + C(16x16 f32)
  return __builtin_amdgcn_wmma_f32_16x16x32_f16(
      /*neg_a=*/false, a, /*neg_b=*/false, b,
      /*c_mod=*/(short)0, c, /*reuse_a=*/false, /*reuse_b=*/false);
}

// --- fragment loaders ------------------------------------------------------
// A layout (ISA 7.12.2, 16-bit A 16x32): lane<16 -> M=lane, K={0..7,16..23};
//                                        lane>=16 -> M=lane-16, K={8..15,24..31}
// B layout (32x16): lane<16 -> N=lane, K={0..7,16..23}; lane>=16 -> N=lane-16, K={8..15,24..31}

static __device__ __forceinline__ v16h lda_f16(const h16* base, int ld) {
  const int lane = threadIdx.x & 31;
  const int l16  = lane & 15;
  const int k0   = (lane & 16) ? 8 : 0;
  const h16* p = base + (size_t)l16 * ld + k0;
  v8h lo = *(const v8h*)(p);
  v8h hi = *(const v8h*)(p + 16);
  return __builtin_shufflevector(lo, hi, 0,1,2,3,4,5,6,7,8,9,10,11,12,13,14,15);
}

static __device__ __forceinline__ v16h ldb_f16(const h16* base, int ld) {
  const int lane = threadIdx.x & 31;
  const int l16  = lane & 15;
  const int k0   = (lane & 16) ? 8 : 0;
  const h16* p = base + (size_t)l16 * ld + k0;
  v8h lo = *(const v8h*)(p);
  v8h hi = *(const v8h*)(p + 16);
  return __builtin_shufflevector(lo, hi, 0,1,2,3,4,5,6,7,8,9,10,11,12,13,14,15);
}

static __device__ __forceinline__ v16h cvt16(v4f f0, v4f f1, v4f f2, v4f f3) {
  v16h r;
#pragma unroll
  for (int i = 0; i < 4; ++i) {
    r[i]      = (h16)f0[i];
    r[4 + i]  = (h16)f1[i];
    r[8 + i]  = (h16)f2[i];
    r[12 + i] = (h16)f3[i];
  }
  return r;
}

static __device__ __forceinline__ v16h lda_f32(const float* base, int ld) {
  const int lane = threadIdx.x & 31;
  const int l16  = lane & 15;
  const int k0   = (lane & 16) ? 8 : 0;
  const float* p = base + (size_t)l16 * ld + k0;
  return cvt16(*(const v4f*)(p),      *(const v4f*)(p + 4),
               *(const v4f*)(p + 16), *(const v4f*)(p + 20));
}

static __device__ __forceinline__ v16h ldb_f32(const float* base, int ld) {
  const int lane = threadIdx.x & 31;
  const int l16  = lane & 15;
  const int k0   = (lane & 16) ? 8 : 0;
  const float* p = base + (size_t)l16 * ld + k0;
  return cvt16(*(const v4f*)(p),      *(const v4f*)(p + 4),
               *(const v4f*)(p + 16), *(const v4f*)(p + 20));
}

// ---------------------------------------------------------------------------
// Kernel 1: QKV GEMM.  qkv[m][d] = sum_k x[m][k] * w_qkv[d][k]
// Block = 128 threads (4 waves), block tile 64x64, wave tile 32x32.
// ---------------------------------------------------------------------------
__global__ __launch_bounds__(128)
void qkv_gemm_kernel(const float* __restrict__ x, const float* __restrict__ wqkv,
                     h16* __restrict__ qws, h16* __restrict__ kws,
                     h16* __restrict__ vtws) {
  const int wave = __builtin_amdgcn_readfirstlane((int)(threadIdx.x >> 5));
  const int wr = wave >> 1, wc = wave & 1;
  const int m0 = blockIdx.y * 64 + wr * 32;   // scalar
  const int n0 = blockIdx.x * 64 + wc * 32;   // scalar

  v8f acc[2][2] = {};
  for (int kk = 0; kk < DIM_C; kk += 32) {
    __builtin_prefetch(x    + (size_t)m0 * DIM_C + kk + 64, 0, 0);
    __builtin_prefetch(wqkv + (size_t)n0 * DIM_C + kk + 64, 0, 0);
    v16h a0 = lda_f32(x + (size_t)m0 * DIM_C + kk, DIM_C);
    v16h a1 = lda_f32(x + (size_t)(m0 + 16) * DIM_C + kk, DIM_C);
    v16h b0 = ldb_f32(wqkv + (size_t)n0 * DIM_C + kk, DIM_C);
    v16h b1 = ldb_f32(wqkv + (size_t)(n0 + 16) * DIM_C + kk, DIM_C);
    acc[0][0] = wmma_f16(a0, b0, acc[0][0]);
    acc[0][1] = wmma_f16(a0, b1, acc[0][1]);
    acc[1][0] = wmma_f16(a1, b0, acc[1][0]);
    acc[1][1] = wmma_f16(a1, b1, acc[1][1]);
  }

  // Epilogue. C layout: lane l, vgpr j -> (row = j + 8*(l>=16), col = l&15).
  // The Q/K/V selector and head index are tile-uniform -> scalar branches.
  const int lane = threadIdx.x & 31;
  const int l16  = lane & 15;
  const int rhi  = (lane & 16) ? 8 : 0;
#pragma unroll
  for (int mi = 0; mi < 2; ++mi) {
    const int gmb = m0 + mi * 16;              // scalar row base
    const int b   = gmb >> 11;                 // scalar batch
    const int nnb = gmb & 2047;                // scalar seq base
#pragma unroll
    for (int ni = 0; ni < 2; ++ni) {
      const int dbase = n0 + ni * 16;          // scalar col base (16-aligned)
      const int s     = dbase >> 10;           // scalar: 0=Q 1=K 2=V
      const int hh    = (dbase & 1023) >> 6;   // scalar head
      const int e     = (dbase & 63) + l16;    // vector head-dim index
      const int bh    = b * NHEADS + hh;       // scalar
#pragma unroll
      for (int j = 0; j < 8; ++j) {
        const int nn = nnb + j + rhi;
        const h16 hv = (h16)acc[mi][ni][j];
        if (s == 0)      qws [((size_t)bh * SEQ + nn) * HD + e] = hv;
        else if (s == 1) kws [((size_t)bh * SEQ + nn) * HD + e] = hv;
        else             vtws[((size_t)bh * HD + e) * SEQ + nn] = hv;
      }
    }
  }
}

// ---------------------------------------------------------------------------
// Kernel 2: fused flash attention. Block = 4 waves; each wave owns 16 Q rows.
// Per 64-key chunk: K tile (64x64) and V tile (64x64, from transposed V) are
// staged block-wide into LDS with async copies; S/P stay in registers, the P
// tile is bounced through per-wave LDS for the C->A layout conversion.
// ---------------------------------------------------------------------------
__global__ __launch_bounds__(128)
void flash_attn_kernel(const h16* __restrict__ qws, const h16* __restrict__ kws,
                       const h16* __restrict__ vtws, h16* __restrict__ aows) {
  __shared__ __align__(16) h16 ktile[64 * 64];     // [key][d]      8 KB
  __shared__ __align__(16) h16 vtile[64 * 64];     // [d][key]      8 KB
  __shared__ __align__(16) h16 pbuf[4][16 * 72];   // per-wave P    9 KB

  const int wave = __builtin_amdgcn_readfirstlane((int)(threadIdx.x >> 5));
  const int lane = threadIdx.x & 31;
  const int l16  = lane & 15;
  const int rhi  = (lane & 16) ? 8 : 0;

  const int bh = blockIdx.y;           // 0..31
  const int b  = bh >> 4, h = bh & 15;
  const int row0 = blockIdx.x * 64 + wave * 16;    // scalar

  const h16* qb = qws  + (size_t)bh * SEQ * HD;
  const h16* kb = kws  + (size_t)bh * SEQ * HD;
  const h16* vb = vtws + (size_t)bh * HD * SEQ;
  h16* pb = &pbuf[wave][0];

  // staging assignment: 2 threads per row, 32 halves (4x b128) each
  const int srow = (int)(threadIdx.x >> 1);
  const int scol = (int)(threadIdx.x & 1) * 32;

  v16h qf0 = lda_f16(qb + (size_t)row0 * HD + 0,  HD);
  v16h qf1 = lda_f16(qb + (size_t)row0 * HD + 32, HD);
  qf0 = qf0 * (h16)QK_SCALE;   // exact pow2 scale folded into Q
  qf1 = qf1 * (h16)QK_SCALE;

  float m[8], l[8];
  v8f o[4] = {};
#pragma unroll
  for (int j = 0; j < 8; ++j) { m[j] = -1.0e30f; l[j] = 0.0f; }

  for (int kc = 0; kc < SEQ; kc += 64) {
    // ---- stage K and V chunk tiles into LDS (async, block cooperative) -----
    __syncthreads();   // previous chunk's readers done before overwrite
#pragma unroll
    for (int i = 0; i < 4; ++i) {
      async_cp_b128(kb + (size_t)(kc + srow) * HD + scol + i * 8,
                    ktile + (size_t)srow * 64 + scol + i * 8);
      async_cp_b128(vb + (size_t)srow * SEQ + kc + scol + i * 8,
                    vtile + (size_t)srow * 64 + scol + i * 8);
    }
    async_wait_all();
    __syncthreads();

    // ---- S = (Q*scale) K^T for 16 rows x 64 key-cols -----------------------
    v8f sfr[4];
#pragma unroll
    for (int jf = 0; jf < 4; ++jf) {
      v8f z = {};
      v16h kf0 = ldb_f16(ktile + (size_t)(jf * 16) * 64 + 0,  64);
      v16h kf1 = ldb_f16(ktile + (size_t)(jf * 16) * 64 + 32, 64);
      z = wmma_f16(qf0, kf0, z);
      z = wmma_f16(qf1, kf1, z);
      sfr[jf] = z;
    }
    // ---- online softmax: row max over 4 frags + 16 lanes -------------------
    float sc[8];
#pragma unroll
    for (int j = 0; j < 8; ++j) {
      float v = sfr[0][j];
      v = fmaxf(v, sfr[1][j]); v = fmaxf(v, sfr[2][j]); v = fmaxf(v, sfr[3][j]);
      v = fmaxf(v, __shfl_xor(v, 1, 16));
      v = fmaxf(v, __shfl_xor(v, 2, 16));
      v = fmaxf(v, __shfl_xor(v, 4, 16));
      v = fmaxf(v, __shfl_xor(v, 8, 16));
      const float mn = fmaxf(m[j], v);
      sc[j] = __expf(m[j] - mn);
      m[j] = mn;
    }
    // ---- P = exp(S - m), row sums, stage P into per-wave LDS ---------------
    float rsum[8];
#pragma unroll
    for (int j = 0; j < 8; ++j) rsum[j] = 0.0f;
#pragma unroll
    for (int jf = 0; jf < 4; ++jf)
#pragma unroll
      for (int j = 0; j < 8; ++j) {
        const float p = __expf(sfr[jf][j] - m[j]);
        rsum[j] += p;
        pb[(size_t)(j + rhi) * 72 + jf * 16 + l16] = (h16)p;
      }
#pragma unroll
    for (int j = 0; j < 8; ++j) {
      float v = rsum[j];
      v += __shfl_xor(v, 1, 16);
      v += __shfl_xor(v, 2, 16);
      v += __shfl_xor(v, 4, 16);
      v += __shfl_xor(v, 8, 16);
      l[j] = l[j] * sc[j] + v;
    }
#pragma unroll
    for (int df = 0; df < 4; ++df)
#pragma unroll
      for (int j = 0; j < 8; ++j) o[df][j] *= sc[j];

    // ---- O += P * V  (P reloaded from LDS in A layout; V tile in LDS) ------
    v16h pf0 = lda_f16(pb + 0,  72);
    v16h pf1 = lda_f16(pb + 32, 72);
#pragma unroll
    for (int df = 0; df < 4; ++df) {
      v16h vf0 = ldb_f16(vtile + (size_t)(df * 16) * 64 + 0,  64);
      v16h vf1 = ldb_f16(vtile + (size_t)(df * 16) * 64 + 32, 64);
      o[df] = wmma_f16(pf0, vf0, o[df]);
      o[df] = wmma_f16(pf1, vf1, o[df]);
    }
  }

  // ---- normalize and write f16 to [b*SEQ+n][h*64+e] ------------------------
#pragma unroll
  for (int j = 0; j < 8; ++j) {
    const float inv = 1.0f / l[j];
    const int nn = row0 + j + rhi;
#pragma unroll
    for (int df = 0; df < 4; ++df) {
      const float v = o[df][j] * inv;
      aows[((size_t)(b * SEQ + nn)) * DIM_C + h * HD + df * 16 + l16] = (h16)v;
    }
  }
}

// ---------------------------------------------------------------------------
// Kernel 3: projection GEMM + bias.  out[m][d] = sum_k ao[m][k]*w_prj[d][k]+b[d]
// ---------------------------------------------------------------------------
__global__ __launch_bounds__(128)
void proj_gemm_kernel(const h16* __restrict__ ao, const float* __restrict__ wprj,
                      const float* __restrict__ bprj, float* __restrict__ out) {
  const int wave = __builtin_amdgcn_readfirstlane((int)(threadIdx.x >> 5));
  const int wr = wave >> 1, wc = wave & 1;
  const int m0 = blockIdx.y * 64 + wr * 32;
  const int n0 = blockIdx.x * 64 + wc * 32;

  v8f acc[2][2] = {};
  for (int kk = 0; kk < DIM_C; kk += 32) {
    __builtin_prefetch(ao   + (size_t)m0 * DIM_C + kk + 64, 0, 0);
    __builtin_prefetch(wprj + (size_t)n0 * DIM_C + kk + 64, 0, 0);
    v16h a0 = lda_f16(ao + (size_t)m0 * DIM_C + kk, DIM_C);
    v16h a1 = lda_f16(ao + (size_t)(m0 + 16) * DIM_C + kk, DIM_C);
    v16h b0 = ldb_f32(wprj + (size_t)n0 * DIM_C + kk, DIM_C);
    v16h b1 = ldb_f32(wprj + (size_t)(n0 + 16) * DIM_C + kk, DIM_C);
    acc[0][0] = wmma_f16(a0, b0, acc[0][0]);
    acc[0][1] = wmma_f16(a0, b1, acc[0][1]);
    acc[1][0] = wmma_f16(a1, b0, acc[1][0]);
    acc[1][1] = wmma_f16(a1, b1, acc[1][1]);
  }

  const int lane = threadIdx.x & 31;
  const int l16  = lane & 15;
  const int rhi  = (lane & 16) ? 8 : 0;
#pragma unroll
  for (int mi = 0; mi < 2; ++mi)
#pragma unroll
    for (int ni = 0; ni < 2; ++ni) {
      const int col = n0 + ni * 16 + l16;
      const float bias = bprj[col];
#pragma unroll
      for (int j = 0; j < 8; ++j) {
        const int gm = m0 + mi * 16 + j + rhi;
        out[(size_t)gm * DIM_C + col] = acc[mi][ni][j] + bias;
      }
    }
}

// ---------------------------------------------------------------------------
extern "C" void kernel_launch(void* const* d_in, const int* in_sizes, int n_in,
                              void* d_out, int out_size, void* d_ws, size_t ws_size,
                              hipStream_t stream) {
  (void)in_sizes; (void)n_in; (void)out_size; (void)ws_size;
  const float* x    = (const float*)d_in[0];
  const float* wqkv = (const float*)d_in[1];
  const float* wprj = (const float*)d_in[2];
  const float* bprj = (const float*)d_in[3];
  float* out = (float*)d_out;

  const size_t HSEG = (size_t)BATCH * NHEADS * SEQ * HD;  // 4,194,304 halves
  h16* qws  = (h16*)d_ws;
  h16* kws  = qws  + HSEG;
  h16* vtws = kws  + HSEG;
  h16* aows = vtws + HSEG;                                // [4096][1024] halves

  const dim3 blk(128, 1, 1);
  qkv_gemm_kernel <<<dim3(3072 / 64, MTOT / 64, 1), blk, 0, stream>>>(x, wqkv, qws, kws, vtws);
  flash_attn_kernel<<<dim3(SEQ / 64, BATCH * NHEADS, 1), blk, 0, stream>>>(qws, kws, vtws, aows);
  proj_gemm_kernel <<<dim3(DIM_C / 64, MTOT / 64, 1), blk, 0, stream>>>(aows, wprj, bprj, out);
}
